// SepViT_86552180949766
// MI455X (gfx1250) — matrix operand order repaired and checked by reading, not compiled, not run on hardware
//
#include <hip/hip_runtime.h>
#include <hip/hip_bf16.h>
#include <math.h>

typedef _Float16 h16;
typedef __attribute__((ext_vector_type(16))) _Float16 v16h;
typedef __attribute__((ext_vector_type(8)))  float    v8f;

// ---- problem constants (fixed by setup_inputs) ----
static constexpr int B_    = 32;
static constexpr int WIN   = 64;       // windows per image
static constexpr int NTOK  = 50;       // tokens per window (1 win_tok + 49)
static constexpr int C_    = 384;
static constexpr int H_    = 12;       // heads
static constexpr int D_    = 32;       // head dim
static constexpr float SCALE = 0.17677669529663687f; // 32^-0.5
static constexpr int BW    = B_ * WIN;             // 2048
static constexpr int M_QKV = BW * NTOK;            // 102400 rows
static constexpr int M_OUT = B_ * 56 * 56;         // 100352 rows

// ------------------------------------------------------------------
// WMMA helpers (wave32, v_wmma_f32_16x16x32_f16)
// ------------------------------------------------------------------
__device__ __forceinline__ v8f wmma16(v16h a, v16h b, v8f c) {
  return __builtin_amdgcn_wmma_f32_16x16x32_f16(false, a, false, b, (short)0, c, false, false);
}

// A fragment 16x32 f16: lane row = lane&15, kq = lane>>4.
// elems 0..15 -> K = kq*16 + e
__device__ __forceinline__ v16h load_a16(const h16* __restrict__ row, int kq) {
  v16h r;
  const h16* p0 = row + kq * 16;
  const h16* p1 = row + kq * 16 + 8;
#pragma unroll
  for (int i = 0; i < 8; ++i) r[i] = p0[i];
#pragma unroll
  for (int i = 0; i < 8; ++i) r[8 + i] = p1[i];
  return r;
}
__device__ __forceinline__ v16h load_a32(const float* __restrict__ row, int kq) {
  v16h r;
  const float* p0 = row + kq * 16;
  const float* p1 = row + kq * 16 + 8;
#pragma unroll
  for (int i = 0; i < 8; ++i) r[i] = (h16)p0[i];
#pragma unroll
  for (int i = 0; i < 8; ++i) r[8 + i] = (h16)p1[i];
  return r;
}
// B fragment 32x16: col = lane&15, elem e = B[kq*16 + e][col].
// For X@W^T, B[k][n] = W[n][k] -> 16 contiguous halves of W row `col`.
__device__ __forceinline__ v16h load_b16(const h16* __restrict__ colrow, int kq) {
  v16h r;
  const h16* p = colrow + kq * 16;
#pragma unroll
  for (int e = 0; e < 16; ++e) r[e] = p[e];
  return r;
}

// ------------------------------------------------------------------
// K1: fp32 -> fp16 convert (weights)
// ------------------------------------------------------------------
__global__ void cvt_f32_f16_kernel(const float* __restrict__ src, h16* __restrict__ dst, int n) {
  int i = blockIdx.x * blockDim.x + threadIdx.x;
  for (; i < n; i += gridDim.x * blockDim.x) dst[i] = (h16)src[i];
}

// ------------------------------------------------------------------
// K2: QKV GEMM  (102400 x 384) @ (1152 x 384)^T, scatter q/k/vT f16
// Register-blocked: 64x32 output per wave (4x2 WMMA sub-tiles).
// tilesM64 = 1600, tilesN32 = 36 -> 57600 waves -> 7200 blocks of 8 waves
// ------------------------------------------------------------------
__global__ void qkv_gemm_kernel(const float* __restrict__ x, const h16* __restrict__ wq,
                                h16* __restrict__ qb, h16* __restrict__ kb,
                                h16* __restrict__ vTb) {
  const int lane = threadIdx.x & 31, wave = threadIdx.x >> 5;
  const int lrow = lane & 15, kq = lane >> 4;
  const int tile = blockIdx.x * 8 + wave;
  const int mt = tile / 36, nt = tile % 36;      // 64-row, 32-col macro tile
  const float* arow[4];
#pragma unroll
  for (int m = 0; m < 4; ++m) arow[m] = x + (size_t)(mt * 64 + m * 16 + lrow) * C_;
  const h16* bcol[2];
#pragma unroll
  for (int n = 0; n < 2; ++n) bcol[n] = wq + (size_t)(nt * 32 + n * 16 + lrow) * C_;

  v8f acc[4][2] = {};
#pragma unroll 2
  for (int kk = 0; kk < 12; ++kk) {
    v16h a[4], b[2];
#pragma unroll
    for (int m = 0; m < 4; ++m) a[m] = load_a32(arow[m] + kk * 32, kq);
#pragma unroll
    for (int n = 0; n < 2; ++n) b[n] = load_b16(bcol[n] + kk * 32, kq);
#pragma unroll
    for (int m = 0; m < 4; ++m)
#pragma unroll
      for (int n = 0; n < 2; ++n) acc[m][n] = wmma16(a[m], b[n], acc[m][n]);
  }

  const int nbase = nt * 32;
  const int sel = nbase / C_;                    // uniform within 32-wide block
  const int head = (nbase % C_) >> 5;            // 32-wide block == one head
#pragma unroll
  for (int n = 0; n < 2; ++n) {
    const int dd = n * 16 + lrow;
#pragma unroll
    for (int m = 0; m < 4; ++m) {
#pragma unroll
      for (int v = 0; v < 8; ++v) {
        const int row = mt * 64 + m * 16 + kq * 8 + v;
        const int bw = row / NTOK, tok = row % NTOK;
        const float val = acc[m][n][v];
        if (sel == 0)
          qb[((size_t)(bw * H_ + head) * NTOK + tok) * D_ + dd] = (h16)(val * SCALE);
        else if (sel == 1)
          kb[((size_t)(bw * H_ + head) * NTOK + tok) * D_ + dd] = (h16)val;
        else
          vTb[((size_t)(bw * H_ + head) * D_ + dd) * 64 + tok] = (h16)val;
      }
    }
  }
}

// ------------------------------------------------------------------
// K3: window attention. grid (2048, 12), block 128 (4 waves, 16-row strips)
// ------------------------------------------------------------------
__global__ void win_attn_kernel(const h16* __restrict__ qb, const h16* __restrict__ kb,
                                const h16* __restrict__ vTb, const float* __restrict__ x,
                                float* __restrict__ outb) {
  __shared__ h16 ldsP[4][16][64];
  const int lane = threadIdx.x & 31, wave = threadIdx.x >> 5;
  const int lrow = lane & 15, kq = lane >> 4;
  const int bw = blockIdx.x, head = blockIdx.y;
  const size_t base = (size_t)(bw * H_ + head) * NTOK * D_;

  int qrow = wave * 16 + lrow;
  qrow = qrow > 49 ? 49 : qrow;              // clamp (padded rows discarded later)
  const v16h aq = load_a16(qb + base + (size_t)qrow * D_, kq);

  v8f s[4];
#pragma unroll
  for (int j = 0; j < 4; ++j) {
    int ct = j * 16 + lrow;
    ct = ct > 49 ? 49 : ct;                  // clamp (masked below)
    v16h bk = load_b16(kb + base + (size_t)ct * D_, kq);
    v8f z = {};
    s[j] = wmma16(aq, bk, z);
  }
  // masked softmax over 64 cols (valid: 50), rows live in 16-lane groups
#pragma unroll
  for (int v = 0; v < 8; ++v) {
    float mx = -INFINITY;
#pragma unroll
    for (int j = 0; j < 4; ++j) {
      float sv = (j * 16 + lrow) < NTOK ? s[j][v] : -INFINITY;
      s[j][v] = sv;
      mx = fmaxf(mx, sv);
    }
    mx = fmaxf(mx, __shfl_xor(mx, 1, 32));
    mx = fmaxf(mx, __shfl_xor(mx, 2, 32));
    mx = fmaxf(mx, __shfl_xor(mx, 4, 32));
    mx = fmaxf(mx, __shfl_xor(mx, 8, 32));
    float e[4], sum = 0.f;
#pragma unroll
    for (int j = 0; j < 4; ++j) { e[j] = __expf(s[j][v] - mx); sum += e[j]; }
    sum += __shfl_xor(sum, 1, 32);
    sum += __shfl_xor(sum, 2, 32);
    sum += __shfl_xor(sum, 4, 32);
    sum += __shfl_xor(sum, 8, 32);
    const float inv = 1.0f / sum;
#pragma unroll
    for (int j = 0; j < 4; ++j)
      ldsP[wave][kq * 8 + v][j * 16 + lrow] = (h16)(e[j] * inv);
  }
  asm volatile("s_wait_dscnt 0" ::: "memory");   // cross-lane LDS RAW, wave lockstep

  const h16* prow = &ldsP[wave][lrow][0];
  const v16h a0 = load_a16(prow, kq);
  const v16h a1 = load_a16(prow + 32, kq);
  const size_t vbase = (size_t)(bw * H_ + head) * D_ * 64;
#pragma unroll
  for (int no = 0; no < 2; ++no) {
    const h16* vrow = vTb + vbase + (size_t)(no * 16 + lrow) * 64;
    v8f acc = {};
    acc = wmma16(a0, load_b16(vrow, kq), acc);
    acc = wmma16(a1, load_b16(vrow + 32, kq), acc);
#pragma unroll
    for (int v = 0; v < 8; ++v) {
      const int t = wave * 16 + kq * 8 + v;
      if (t < NTOK) {
        const size_t idx = ((size_t)bw * NTOK + t) * C_ + head * D_ + no * 16 + lrow;
        outb[idx] = acc[v] + x[idx];
      }
    }
  }
}

// ------------------------------------------------------------------
// K4: win_tok LayerNorm + exact GeLU -> f16. one wave per window token
// ------------------------------------------------------------------
__global__ void ln_gelu_kernel(const float* __restrict__ outb, const float* __restrict__ g,
                               const float* __restrict__ bta, h16* __restrict__ wt) {
  const int lane = threadIdx.x & 31, wave = threadIdx.x >> 5;
  const int tokw = blockIdx.x * 8 + wave;              // 0..2047
  const float* row = outb + (size_t)tokw * NTOK * C_;  // token 0 of this window
  float s1 = 0.f, s2 = 0.f;
  for (int c = lane; c < C_; c += 32) { float v = row[c]; s1 += v; s2 += v * v; }
  s1 += __shfl_xor(s1, 1, 32);  s2 += __shfl_xor(s2, 1, 32);
  s1 += __shfl_xor(s1, 2, 32);  s2 += __shfl_xor(s2, 2, 32);
  s1 += __shfl_xor(s1, 4, 32);  s2 += __shfl_xor(s2, 4, 32);
  s1 += __shfl_xor(s1, 8, 32);  s2 += __shfl_xor(s2, 8, 32);
  s1 += __shfl_xor(s1, 16, 32); s2 += __shfl_xor(s2, 16, 32);
  const float mean = s1 / (float)C_;
  const float var = s2 / (float)C_ - mean * mean;
  const float inv = rsqrtf(var + 1e-5f);
  for (int c = lane; c < C_; c += 32) {
    float v = (row[c] - mean) * inv * g[c] + bta[c];
    float ge = 0.5f * v * (1.0f + erff(v * 0.70710678118654752f));
    wt[(size_t)tokw * C_ + c] = (h16)ge;
  }
}

// ------------------------------------------------------------------
// K5: pqk GEMM (2048 x 384) @ (768 x 384)^T -> pq (scaled), pk f16
// tilesM=128, tilesN=48
// ------------------------------------------------------------------
__global__ void pqk_gemm_kernel(const h16* __restrict__ wt, const h16* __restrict__ wqk,
                                h16* __restrict__ pq, h16* __restrict__ pk) {
  const int lane = threadIdx.x & 31, wave = threadIdx.x >> 5;
  const int lrow = lane & 15, kq = lane >> 4;
  const int tile = blockIdx.x * 8 + wave;
  const int mt = tile / 48, nt = tile % 48;
  const h16* arow = wt + (size_t)(mt * 16 + lrow) * C_;
  const h16* bcol = wqk + (size_t)(nt * 16 + lrow) * C_;
  v8f acc = {};
#pragma unroll 4
  for (int kk = 0; kk < 12; ++kk)
    acc = wmma16(load_a16(arow + kk * 32, kq), load_b16(bcol + kk * 32, kq), acc);
  const int nGlob = nt * 16 + lrow;
  const int sel = nGlob / C_, c = nGlob % C_;
#pragma unroll
  for (int v = 0; v < 8; ++v) {
    const int m = mt * 16 + kq * 8 + v;
    if (sel == 0) pq[(size_t)m * C_ + c] = (h16)(acc[v] * SCALE);
    else          pk[(size_t)m * C_ + c] = (h16)acc[v];
  }
}

// ------------------------------------------------------------------
// K6: window-token attention map. grid 384 = (b,head), block 128
// ------------------------------------------------------------------
__global__ void pattn_kernel(const h16* __restrict__ pq, const h16* __restrict__ pk,
                             h16* __restrict__ pattn) {
  const int lane = threadIdx.x & 31, wave = threadIdx.x >> 5;
  const int lrow = lane & 15, kq = lane >> 4;
  const int b = blockIdx.x / H_, head = blockIdx.x % H_;
  const v16h a = load_a16(pq + ((size_t)(b * WIN + wave * 16 + lrow)) * C_ + head * D_, kq);
  v8f s[4];
#pragma unroll
  for (int j = 0; j < 4; ++j) {
    v16h bf = load_b16(pk + ((size_t)(b * WIN + j * 16 + lrow)) * C_ + head * D_, kq);
    v8f z = {};
    s[j] = wmma16(a, bf, z);
  }
#pragma unroll
  for (int v = 0; v < 8; ++v) {
    float mx = -INFINITY;
#pragma unroll
    for (int j = 0; j < 4; ++j) mx = fmaxf(mx, s[j][v]);
    mx = fmaxf(mx, __shfl_xor(mx, 1, 32));
    mx = fmaxf(mx, __shfl_xor(mx, 2, 32));
    mx = fmaxf(mx, __shfl_xor(mx, 4, 32));
    mx = fmaxf(mx, __shfl_xor(mx, 8, 32));
    float e[4], sum = 0.f;
#pragma unroll
    for (int j = 0; j < 4; ++j) { e[j] = __expf(s[j][v] - mx); sum += e[j]; }
    sum += __shfl_xor(sum, 1, 32);
    sum += __shfl_xor(sum, 2, 32);
    sum += __shfl_xor(sum, 4, 32);
    sum += __shfl_xor(sum, 8, 32);
    const float inv = 1.0f / sum;
    const int row = wave * 16 + kq * 8 + v;
#pragma unroll
    for (int j = 0; j < 4; ++j)
      pattn[((size_t)blockIdx.x * WIN + row) * WIN + j * 16 + lrow] = (h16)(e[j] * inv);
  }
}

// ------------------------------------------------------------------
// K7: psa = pattn @ ax  + residual, window un-shuffle, -> y f16
// grid (98, 384): x = N tile (1568/16), y = (b,head); wave = M tile
// ------------------------------------------------------------------
__global__ void psa_kernel(const h16* __restrict__ pattn, const float* __restrict__ outb,
                           h16* __restrict__ yh) {
  const int lane = threadIdx.x & 31, wave = threadIdx.x >> 5;
  const int lrow = lane & 15, kq = lane >> 4;
  const int nt = blockIdx.x;
  const int bh = blockIdx.y, b = bh / H_, head = bh % H_;
  const int mt = wave;
  const h16* arow = pattn + ((size_t)bh * WIN + mt * 16 + lrow) * WIN;
  const int n = nt * 16 + lrow;                 // 0..1567
  const int s = n >> 5, dd = n & 31;            // s: 0..48
  v8f acc = {};
#pragma unroll
  for (int kk = 0; kk < 2; ++kk) {
    v16h a = load_a16(arow + kk * 32, kq);
    v16h bf;
#pragma unroll
    for (int e = 0; e < 16; ++e) {
      const int vwin = kk * 32 + kq * 16 + e;   // window index (K dim)
      bf[e] = (h16)outb[((size_t)(b * WIN + vwin) * NTOK + 1 + s) * C_ + head * D_ + dd];
    }
    acc = wmma16(a, bf, acc);
  }
  const int rr = s / 7, cc = s % 7;
#pragma unroll
  for (int v = 0; v < 8; ++v) {
    const int w = mt * 16 + kq * 8 + v;         // window (output row)
    const float res = outb[((size_t)(b * WIN + w) * NTOK + 1 + s) * C_ + head * D_ + dd];
    const float y = acc[v] + res;
    const int hh = w >> 3, ww = w & 7;
    const int py = hh * 7 + rr, px = ww * 7 + cc;
    yh[((size_t)b * 3136 + py * 56 + px) * C_ + head * D_ + dd] = (h16)y;
  }
}

// ------------------------------------------------------------------
// K8: proj GEMM (100352 x 384) @ (384 x 384)^T + bias -> d_out f32
// Register-blocked 64x32 per wave: tilesM64=1568, tilesN32=12 -> 2352 blocks
// ------------------------------------------------------------------
__global__ void proj_gemm_kernel(const h16* __restrict__ yh, const h16* __restrict__ wp,
                                 const float* __restrict__ bias, float* __restrict__ out) {
  const int lane = threadIdx.x & 31, wave = threadIdx.x >> 5;
  const int lrow = lane & 15, kq = lane >> 4;
  const int tile = blockIdx.x * 8 + wave;
  const int mt = tile / 12, nt = tile % 12;
  const h16* arow[4];
#pragma unroll
  for (int m = 0; m < 4; ++m) arow[m] = yh + (size_t)(mt * 64 + m * 16 + lrow) * C_;
  const h16* bcol[2];
#pragma unroll
  for (int n = 0; n < 2; ++n) bcol[n] = wp + (size_t)(nt * 32 + n * 16 + lrow) * C_;

  v8f acc[4][2] = {};
#pragma unroll 2
  for (int kk = 0; kk < 12; ++kk) {
    v16h a[4], b[2];
#pragma unroll
    for (int m = 0; m < 4; ++m) a[m] = load_a16(arow[m] + kk * 32, kq);
#pragma unroll
    for (int n = 0; n < 2; ++n) b[n] = load_b16(bcol[n] + kk * 32, kq);
#pragma unroll
    for (int m = 0; m < 4; ++m)
#pragma unroll
      for (int n = 0; n < 2; ++n) acc[m][n] = wmma16(a[m], b[n], acc[m][n]);
  }
#pragma unroll
  for (int n = 0; n < 2; ++n) {
    const int nGlob = nt * 32 + n * 16 + lrow;
    const float bn = bias[nGlob];
#pragma unroll
    for (int m = 0; m < 4; ++m)
#pragma unroll
      for (int v = 0; v < 8; ++v)
        out[(size_t)(mt * 64 + m * 16 + kq * 8 + v) * C_ + nGlob] = acc[m][n][v] + bn;
  }
}

// ------------------------------------------------------------------
extern "C" void kernel_launch(void* const* d_in, const int* in_sizes, int n_in,
                              void* d_out, int out_size, void* d_ws, size_t ws_size,
                              hipStream_t stream) {
  const float* x      = (const float*)d_in[0];
  const float* w_qkv  = (const float*)d_in[3];
  const float* w_qk   = (const float*)d_in[4];
  const float* ln_g   = (const float*)d_in[5];
  const float* ln_b   = (const float*)d_in[6];
  const float* w_proj = (const float*)d_in[7];
  const float* b_proj = (const float*)d_in[8];
  float* out = (float*)d_out;

  char* ws = (char*)d_ws;
  size_t off = 0;
  auto take = [&](size_t bytes) -> char* {
    char* p = ws + off;
    off = (off + bytes + 255) & ~(size_t)255;
    return p;
  };
  const size_t QK_BYTES = (size_t)BW * H_ * NTOK * D_ * sizeof(h16);   // 78.6 MB
  h16* q_h  = (h16*)take(QK_BYTES);                    // reused as y_h after K3
  char* kreg = take(QK_BYTES);                         // k, later small buffers
  h16* k_h  = (h16*)kreg;
  h16* vT_h = (h16*)take((size_t)BW * H_ * D_ * 64 * sizeof(h16));     // 100.7 MB
  float* out_f = (float*)take((size_t)M_QKV * C_ * sizeof(float));     // 157.3 MB
  h16* wqkv_h = (h16*)take((size_t)3 * C_ * C_ * sizeof(h16));
  h16* wqk_h  = (h16*)take((size_t)2 * C_ * C_ * sizeof(h16));
  h16* wproj_h= (h16*)take((size_t)C_ * C_ * sizeof(h16));
  // small buffers alias the k region (k dead after win_attn; written after)
  h16* wt_h    = (h16*)(kreg);                                  // 2048*384
  h16* pq_h    = (h16*)(kreg + ((size_t)2048 * C_ * 2 + 256));
  h16* pk_h    = (h16*)(kreg + ((size_t)2 * 2048 * C_ * 2 + 512));
  h16* pattn_h = (h16*)(kreg + ((size_t)3 * 2048 * C_ * 2 + 768)); // 384*64*64
  h16* y_h     = q_h;                                           // q dead after win_attn

  // K1: weight conversions
  cvt_f32_f16_kernel<<<1024, 256, 0, stream>>>(w_qkv, wqkv_h, 3 * C_ * C_);
  cvt_f32_f16_kernel<<<1024, 256, 0, stream>>>(w_qk, wqk_h, 2 * C_ * C_);
  cvt_f32_f16_kernel<<<576, 256, 0, stream>>>(w_proj, wproj_h, C_ * C_);

  // K2: QKV GEMM -> q/k/vT   (1600*36 macro tiles / 8 waves)
  qkv_gemm_kernel<<<7200, 256, 0, stream>>>(x, wqkv_h, q_h, k_h, vT_h);

  // K3: window attention + residual -> out_f
  win_attn_kernel<<<dim3(BW, H_), 128, 0, stream>>>(q_h, k_h, vT_h, x, out_f);

  // K4: LN + GeLU on win_tok -> wt_h
  ln_gelu_kernel<<<256, 256, 0, stream>>>(out_f, ln_g, ln_b, wt_h);

  // K5: pqk GEMM (128*48 tiles / 8)
  pqk_gemm_kernel<<<768, 256, 0, stream>>>(wt_h, wqk_h, pq_h, pk_h);

  // K6: window-token attention map
  pattn_kernel<<<B_ * H_, 128, 0, stream>>>(pq_h, pk_h, pattn_h);

  // K7: psa + residual + window un-shuffle -> y_h
  psa_kernel<<<dim3(98, B_ * H_), 128, 0, stream>>>(pattn_h, out_f, y_h);

  // K8: output projection (1568*12 macro tiles / 8)
  proj_gemm_kernel<<<2352, 256, 0, stream>>>(y_h, wproj_h, b_proj, out);
}
